// TransformerBlock_49108656063286
// MI455X (gfx1250) — compile-verified, matrix operand order checked by reading, not run on hardware
//
#include <hip/hip_runtime.h>
#include <hip/hip_bf16.h>
#include <math.h>

// ---------------------------------------------------------------------------
// Types for CDNA5 WMMA
// ---------------------------------------------------------------------------
typedef __attribute__((ext_vector_type(16))) __bf16 v16bf;
typedef __attribute__((ext_vector_type(8)))  float  v8f;

union FragBF16 {
    v16bf v;
    unsigned short us[16];
    unsigned int   u32[8];
    uint4          q[2];
};

__device__ __forceinline__ unsigned short f2bf(float f) {
    union { float f; unsigned u; } x; x.f = f;
    unsigned u = x.u + 0x7FFFu + ((x.u >> 16) & 1u);   // round-to-nearest-even
    return (unsigned short)(u >> 16);
}

__device__ __forceinline__ unsigned pack2(float a, float b) {
    return (unsigned)f2bf(a) | ((unsigned)f2bf(b) << 16);
}

// 16-bit A/B fragment layout (16x32 / 32x16):
//   elements 0..7  <-> K =  0..7  + (half ? 8 : 0)
//   elements 8..15 <-> K = 16..23 + (half ? 8 : 0)
// => fragment = two contiguous 16-byte LDS reads.
__device__ __forceinline__ void load_frag(FragBF16& f, const unsigned short* row,
                                          int kofs, int half) {
    f.q[0] = *(const uint4*)(row + kofs + half * 8);
    f.q[1] = *(const uint4*)(row + kofs + 16 + half * 8);
}

// Max-reduce across the 16 lanes of a row half using v_permlane16_b32 (VALU),
// avoiding ds_bpermute traffic.
__device__ __forceinline__ float rowmax16(float v) {
#if __has_builtin(__builtin_amdgcn_permlane16)
    #define PL16_STEP(s0, s1)                                                   \
        {                                                                       \
            int iv = __builtin_bit_cast(int, v);                                \
            int rp = __builtin_amdgcn_permlane16(iv, iv, (int)(s0), (int)(s1),  \
                                                 false, false);                 \
            v = fmaxf(v, __builtin_bit_cast(float, rp));                        \
        }
    PL16_STEP(0x67452301u, 0xEFCDAB89u)   // xor 1
    PL16_STEP(0x54761032u, 0xDCFE98BAu)   // xor 2
    PL16_STEP(0x32107654u, 0xBA98FEDCu)   // xor 4
    PL16_STEP(0xFEDCBA98u, 0x76543210u)   // xor 8
    #undef PL16_STEP
#else
    #pragma unroll
    for (int msk = 8; msk; msk >>= 1)
        v = fmaxf(v, __shfl_xor(v, msk, 32));
#endif
    return v;
}

// ---------------------------------------------------------------------------
// Generic GEMM: C[M,N] = act(A[M,K] * W[N,K]^T + bias[N])
// A, W fp32 in global; converted to bf16 while staging into LDS.
// Block tile 128(M) x 64(N), K-step 32; 8 waves in 4x2 grid, 32x32 per wave.
// LDS row stride 40 ushorts (80 B): 16B-aligned rows, conflict-free b128 reads.
// ACT: 0 = none, 1 = exact-erf GELU
// ---------------------------------------------------------------------------
template <int ACT>
__global__ __launch_bounds__(256) void gemm_bias_kernel(
    const float* __restrict__ A, const float* __restrict__ W,
    const float* __restrict__ bias, float* __restrict__ C,
    int M, int N, int K)
{
    constexpr int BM = 128, BN = 64, BK = 32, LDT = 40;
    __shared__ alignas(16) unsigned short As[BM][LDT];
    __shared__ alignas(16) unsigned short Ws[BN][LDT];

    const int tid  = threadIdx.x;
    const int lane = tid & 31;
    const int wave = tid >> 5;
    const int half = lane >> 4;
    const int l16  = lane & 15;
    const int wm   = wave >> 1;          // 0..3
    const int wn   = wave & 1;           // 0..1
    const int bm   = blockIdx.y * BM;
    const int bn   = blockIdx.x * BN;

    v8f acc[2][2] = {};

    for (int k0 = 0; k0 < K; k0 += BK) {
        __syncthreads();
        // ---- stage A tile: 128x32 fp32 -> bf16 (4 float4 per thread) ----
        #pragma unroll
        for (int i = 0; i < 4; ++i) {
            int idx = tid + i * 256;                 // 0..1023
            int r   = idx >> 3;                      // 0..127
            int c4  = (idx & 7) << 2;                // 0..28
            const float4 f = *(const float4*)(A + (size_t)(bm + r) * K + k0 + c4);
            *(unsigned*)&As[r][c4]     = pack2(f.x, f.y);
            *(unsigned*)&As[r][c4 + 2] = pack2(f.z, f.w);
        }
        // ---- stage W tile: 64x32 fp32 -> bf16 (2 float4 per thread) ----
        #pragma unroll
        for (int i = 0; i < 2; ++i) {
            int idx = tid + i * 256;                 // 0..511
            int r   = idx >> 3;                      // 0..63
            int c4  = (idx & 7) << 2;
            const float4 f = *(const float4*)(W + (size_t)(bn + r) * K + k0 + c4);
            *(unsigned*)&Ws[r][c4]     = pack2(f.x, f.y);
            *(unsigned*)&Ws[r][c4 + 2] = pack2(f.z, f.w);
        }
        // prefetch next K-tiles toward L2 while we compute
        if (k0 + BK < K) {
            __builtin_prefetch(A + (size_t)(bm + (tid >> 1)) * K + k0 + BK, 0, 1);
            __builtin_prefetch(W + (size_t)(bn + (tid & 63)) * K + k0 + BK, 0, 1);
        }
        __syncthreads();

        // ---- build fragments: 2x ds_load_b128 each ----
        FragBF16 a[2], b[2];
        #pragma unroll
        for (int t = 0; t < 2; ++t) {
            load_frag(a[t], &As[wm * 32 + t * 16 + l16][0], 0, half);
            load_frag(b[t], &Ws[wn * 32 + t * 16 + l16][0], 0, half);
        }
        #pragma unroll
        for (int tm = 0; tm < 2; ++tm)
            #pragma unroll
            for (int tn = 0; tn < 2; ++tn)
                acc[tm][tn] = __builtin_amdgcn_wmma_f32_16x16x32_bf16(
                    false, a[tm].v, false, b[tn].v, (short)0, acc[tm][tn], false, false);
    }

    // ---- epilogue: bias + activation, fp32 store ----
    #pragma unroll
    for (int tm = 0; tm < 2; ++tm) {
        #pragma unroll
        for (int tn = 0; tn < 2; ++tn) {
            int col = bn + wn * 32 + tn * 16 + l16;
            float bv = bias ? bias[col] : 0.0f;
            #pragma unroll
            for (int i = 0; i < 8; ++i) {
                int row = bm + wm * 32 + tm * 16 + half * 8 + i;
                float v = acc[tm][tn][i] + bv;
                if (ACT == 1)
                    v = 0.5f * v * (1.0f + erff(v * 0.70710678118654752f));
                C[(size_t)row * N + col] = v;
            }
        }
    }
}

// ---------------------------------------------------------------------------
// Flash attention for qkv layout [B,S, 3,H,D] (cols: three*E + h*D + d)
// One block = one (b,h) pair x 128 query rows; 8 waves -> 16 rows each.
// Key blocks of 32; scores & PV via v_wmma_f32_16x16x32_bf16.
// V staged TRANSPOSED (Vt[d][kk]) so PV B-fragments are b128 loads too.
// Row-sums computed by an extra WMMA against an all-ones B fragment
// (no cross-lane sum reduction needed); row-max via v_permlane16 butterfly.
// Output written as [B,S,E] with col = h*D + d (ready for the W_out GEMM).
// ---------------------------------------------------------------------------
__global__ __launch_bounds__(256) void attn_kernel(
    const float* __restrict__ qkv, float* __restrict__ out)
{
    constexpr int S = 2048, D = 64, H = 16, E = 1024, R3E = 3072;
    constexpr int LQ = 72;   // Q/K row stride (ushorts): 144 B, 16B-aligned
    constexpr int LP = 40;   // P/Vt row stride (ushorts): 80 B, 16B-aligned
    constexpr float SCALE = 0.125f;   // 1/sqrt(64)

    __shared__ alignas(16) unsigned short Qs[128][LQ];
    __shared__ alignas(16) unsigned short Ks[32][LQ];
    __shared__ alignas(16) unsigned short Vt[64][LP];     // [d][kk]
    __shared__ alignas(16) unsigned short Ps[8][16][LP];

    const int tid   = threadIdx.x;
    const int lane  = tid & 31;
    const int wave  = tid >> 5;
    const int half  = lane >> 4;
    const int l16   = lane & 15;
    const int bh    = blockIdx.y;
    const int b     = bh / H;
    const int h     = bh % H;
    const int srow0 = blockIdx.x * 128;
    const size_t rowbase = (size_t)(b * S + srow0);

    // ---- stage Q block (pre-scaled): 128x64, 8 float4 per thread ----
    #pragma unroll
    for (int i = 0; i < 8; ++i) {
        int idx = tid + i * 256;                 // 0..2047
        int r   = idx >> 4;                      // 0..127
        int c4  = (idx & 15) << 2;               // 0..60
        const float4 f = *(const float4*)(qkv + (rowbase + r) * R3E + h * D + c4);
        *(unsigned*)&Qs[r][c4]     = pack2(f.x * SCALE, f.y * SCALE);
        *(unsigned*)&Qs[r][c4 + 2] = pack2(f.z * SCALE, f.w * SCALE);
    }
    __syncthreads();

    // Q A-fragments for this wave's 16-row strip (loop invariant)
    const int rbase = wave * 16;
    FragBF16 aq[2];
    #pragma unroll
    for (int kt = 0; kt < 2; ++kt)
        load_frag(aq[kt], &Qs[rbase + l16][0], kt * 32, half);

    // all-ones bf16 B-fragment (1.0 = 0x3F80) for WMMA row-sums
    FragBF16 ones;
    #pragma unroll
    for (int p = 0; p < 8; ++p) ones.u32[p] = 0x3F803F80u;

    // o[0..3]: output tiles over D; o[4]: row-sum (l) tile
    v8f o[5] = {};
    float mrow[8];
    #pragma unroll
    for (int i = 0; i < 8; ++i) mrow[i] = -INFINITY;

    for (int j0 = 0; j0 < S; j0 += 32) {
        __syncthreads();
        // ---- stage K (row-major) and V (transposed) blocks, 32x64 each ----
        #pragma unroll
        for (int i = 0; i < 2; ++i) {
            int idx = tid + i * 256;             // 0..511
            int r   = idx >> 4;                  // 0..31 (key index kk)
            int c4  = (idx & 15) << 2;           // 0..60 (d)
            const size_t grow = (size_t)(b * S + j0 + r) * R3E + h * D + c4;
            const float4 fk = *(const float4*)(qkv + grow + E);
            const float4 fv = *(const float4*)(qkv + grow + 2 * E);
            *(unsigned*)&Ks[r][c4]     = pack2(fk.x, fk.y);
            *(unsigned*)&Ks[r][c4 + 2] = pack2(fk.z, fk.w);
            Vt[c4 + 0][r] = f2bf(fv.x);
            Vt[c4 + 1][r] = f2bf(fv.y);
            Vt[c4 + 2][r] = f2bf(fv.z);
            Vt[c4 + 3][r] = f2bf(fv.w);
        }
        __syncthreads();

        // ---- scores: S_tile[16x32] = Q_strip . K_blk^T (contract over D) ----
        v8f sc[2];
        #pragma unroll
        for (int nt = 0; nt < 2; ++nt) {
            v8f s = {};
            #pragma unroll
            for (int kt = 0; kt < 2; ++kt) {
                FragBF16 bk;
                load_frag(bk, &Ks[nt * 16 + l16][0], kt * 32, half);
                s = __builtin_amdgcn_wmma_f32_16x16x32_bf16(
                        false, aq[kt].v, false, bk.v, (short)0, s, false, false);
            }
            sc[nt] = s;
        }

        // ---- online softmax: max via permlane16; sums come from WMMA ----
        #pragma unroll
        for (int i = 0; i < 8; ++i) {
            float mb   = rowmax16(fmaxf(sc[0][i], sc[1][i]));
            float mnew = fmaxf(mrow[i], mb);
            float alpha = __expf(fminf(mrow[i] - mnew, 0.0f));
            float p0 = __expf(sc[0][i] - mnew);
            float p1 = __expf(sc[1][i] - mnew);
            int r = i + half * 8;
            Ps[wave][r][l16]      = f2bf(p0);
            Ps[wave][r][16 + l16] = f2bf(p1);
            mrow[i] = mnew;
            #pragma unroll
            for (int t = 0; t < 5; ++t) o[t][i] *= alpha;
        }

        // ---- re-layout P (C-layout -> A-fragment) via per-wave LDS ----
        FragBF16 ap;
        load_frag(ap, &Ps[wave][l16][0], 0, half);

        // ---- O += P[16x32] . V[32x64]; l += P . 1 ----
        #pragma unroll
        for (int t = 0; t < 4; ++t) {
            FragBF16 bv;
            load_frag(bv, &Vt[t * 16 + l16][0], 0, half);
            o[t] = __builtin_amdgcn_wmma_f32_16x16x32_bf16(
                       false, ap.v, false, bv.v, (short)0, o[t], false, false);
        }
        o[4] = __builtin_amdgcn_wmma_f32_16x16x32_bf16(
                   false, ap.v, false, ones.v, (short)0, o[4], false, false);
    }

    // ---- normalize by l (each lane holds its row's sum) and write out ----
    #pragma unroll
    for (int i = 0; i < 8; ++i) {
        int r = rbase + i + half * 8;
        float l   = o[4][i];
        float inv = (l > 0.0f) ? (1.0f / l) : 0.0f;
        #pragma unroll
        for (int t = 0; t < 4; ++t)
            out[(rowbase + r) * E + h * D + t * 16 + l16] = o[t][i] * inv;
    }
}

// ---------------------------------------------------------------------------
// Fused residual-add + LayerNorm over E=1024 (one row per block, 256 threads)
// ---------------------------------------------------------------------------
__global__ __launch_bounds__(256) void add_ln_kernel(
    const float* __restrict__ x, const float* __restrict__ res,
    const float* __restrict__ g, const float* __restrict__ beta,
    float* __restrict__ out)
{
    __shared__ float red[16];
    const int tid  = threadIdx.x;
    const int lane = tid & 31;
    const int wave = tid >> 5;
    const size_t base = (size_t)blockIdx.x * 1024;

    float v[4], s = 0.0f, s2 = 0.0f;
    #pragma unroll
    for (int k = 0; k < 4; ++k) {
        int c = tid + k * 256;
        float t = x[base + c] + res[base + c];
        v[k] = t; s += t; s2 += t * t;
    }
    #pragma unroll
    for (int msk = 16; msk; msk >>= 1) {
        s  += __shfl_xor(s,  msk, 32);
        s2 += __shfl_xor(s2, msk, 32);
    }
    if (lane == 0) { red[wave] = s; red[8 + wave] = s2; }
    __syncthreads();
    float ts = 0.0f, ts2 = 0.0f;
    #pragma unroll
    for (int w = 0; w < 8; ++w) { ts += red[w]; ts2 += red[8 + w]; }
    float mu   = ts * (1.0f / 1024.0f);
    float var  = ts2 * (1.0f / 1024.0f) - mu * mu;
    float rinv = rsqrtf(var + 1e-5f);
    #pragma unroll
    for (int k = 0; k < 4; ++k) {
        int c = tid + k * 256;
        out[base + c] = (v[k] - mu) * rinv * g[c] + beta[c];
    }
}

// ---------------------------------------------------------------------------
// Launcher
// ---------------------------------------------------------------------------
extern "C" void kernel_launch(void* const* d_in, const int* in_sizes, int n_in,
                              void* d_out, int out_size, void* d_ws, size_t ws_size,
                              hipStream_t stream) {
    (void)in_sizes; (void)n_in; (void)out_size; (void)ws_size;

    const float* x      = (const float*)d_in[0];
    const float* w_qkv  = (const float*)d_in[1];
    const float* b_qkv  = (const float*)d_in[2];
    const float* w_out  = (const float*)d_in[3];
    const float* b_out  = (const float*)d_in[4];
    const float* g1     = (const float*)d_in[5];
    const float* beta1  = (const float*)d_in[6];
    const float* w_fc1  = (const float*)d_in[7];
    const float* b_fc1  = (const float*)d_in[8];
    const float* w_fc2  = (const float*)d_in[9];
    const float* b_fc2  = (const float*)d_in[10];
    const float* g2     = (const float*)d_in[11];
    const float* beta2  = (const float*)d_in[12];
    float* out = (float*)d_out;

    constexpr int B = 2, S = 2048, E = 1024, FF = 4096;
    constexpr int M = B * S;                         // 4096 tokens

    // workspace layout (floats), with liveness-based reuse:
    float* ws    = (float*)d_ws;
    float* qkv   = ws;                               // [M,3E]  48 MB  (dead after attn)
    float* attnO = ws + (size_t)M * 3 * E;           // [M,E]   16 MB  (dead after proj)
    float* projO = ws + (size_t)M * 4 * E;           // [M,E]   16 MB
    float* x1    = ws + (size_t)M * 5 * E;           // [M,E]   16 MB
    float* hbuf  = ws;                               // [M,FF]  64 MB  (reuses qkv+attnO)
    float* ffO   = ws + (size_t)M * 6 * E;           // [M,E]   16 MB

    // 1. QKV projection: [4096,1024] x [3072,1024]^T
    gemm_bias_kernel<0><<<dim3(3 * E / 64, M / 128), 256, 0, stream>>>(
        x, w_qkv, b_qkv, qkv, M, 3 * E, E);

    // 2. Flash attention: grid (S/128, B*H)
    attn_kernel<<<dim3(S / 128, B * 16), 256, 0, stream>>>(qkv, attnO);

    // 3. Output projection
    gemm_bias_kernel<0><<<dim3(E / 64, M / 128), 256, 0, stream>>>(
        attnO, w_out, b_out, projO, M, E, E);

    // 4. x1 = LN(x + attn_out)
    add_ln_kernel<<<M, 256, 0, stream>>>(x, projO, g1, beta1, x1);

    // 5. h = gelu(x1 @ w_fc1^T + b_fc1)
    gemm_bias_kernel<1><<<dim3(FF / 64, M / 128), 256, 0, stream>>>(
        x1, w_fc1, b_fc1, hbuf, M, FF, E);

    // 6. ff_out = h @ w_fc2^T + b_fc2
    gemm_bias_kernel<0><<<dim3(E / 64, M / 128), 256, 0, stream>>>(
        hbuf, w_fc2, b_fc2, ffO, M, E, FF);

    // 7. out = LN(x1 + ff_out)
    add_ln_kernel<<<M, 256, 0, stream>>>(x1, ffO, g2, beta2, out);
}
